// Attention_76055280878095
// MI455X (gfx1250) — compile-verified, matrix operand order checked by reading, not run on hardware
//
#include <hip/hip_runtime.h>
#include <hip/hip_bf16.h>

typedef __bf16 bf16;
typedef bf16  v16bf __attribute__((ext_vector_type(16)));
typedef bf16  v8bf  __attribute__((ext_vector_type(8)));
typedef bf16  v4bf  __attribute__((ext_vector_type(4)));
typedef float v8f   __attribute__((ext_vector_type(8)));
typedef float v4f   __attribute__((ext_vector_type(4)));

#define B_   32
#define T_   2048
#define D1_  512

// ---- branch-free tanh: prefer gfx1250 V_TANH_F32, else rational exp form ----
__device__ __forceinline__ float fast_tanhf(float x) {
#if __has_builtin(__builtin_amdgcn_tanhf)
    return __builtin_amdgcn_tanhf(x);
#elif __has_builtin(__builtin_amdgcn_tanh_f32)
    return __builtin_amdgcn_tanh_f32(x);
#else
    float xc = fminf(fmaxf(x, -15.0f), 15.0f);
    float t  = __expf(2.0f * xc);
    return (t - 1.0f) * __frcp_rn(t + 1.0f);
#endif
}

// ---- lane-xor exchange: single ds_swizzle (group-of-32 bitmask mode) ----
// offset = {xor_mask[4:0]<<10 | or_mask<<5 | and_mask=0x1F}
__device__ __forceinline__ float swz_xor(float x, int s) {
#if __has_builtin(__builtin_amdgcn_ds_swizzle)
    int xi = __float_as_int(x), r;
    switch (s) {
        case 0:  r = __builtin_amdgcn_ds_swizzle(xi, 0x041F); break;  // xor 1
        case 1:  r = __builtin_amdgcn_ds_swizzle(xi, 0x081F); break;  // xor 2
        case 2:  r = __builtin_amdgcn_ds_swizzle(xi, 0x101F); break;  // xor 4
        default: r = __builtin_amdgcn_ds_swizzle(xi, 0x201F); break;  // xor 8
    }
    return __int_as_float(r);
#else
    return __shfl_xor(x, 1 << s, 32);
#endif
}

// ---- order-preserving float <-> uint map (exact, order-invariant atomic max) ----
__device__ __forceinline__ unsigned encf(float f) {
    unsigned u = __float_as_uint(f);
    return (u & 0x80000000u) ? ~u : (u | 0x80000000u);
}
__device__ __forceinline__ float decf(unsigned u) {
    return __uint_as_float((u & 0x80000000u) ? (u & 0x7FFFFFFFu) : ~u);
}

// ---- K0: init encoded max buffer to enc(-inf) = 0x007FFFFF ----
__global__ void k_init(unsigned* __restrict__ penc) {
    int i = blockIdx.x * blockDim.x + threadIdx.x;
    if (i < B_ * D1_) penc[i] = 0x007FFFFFu;
}

// ---- K1: masked max-pool over T via uint atomicMax ----
__global__ void k_maxpool(const float* __restrict__ x, const int* __restrict__ mask,
                          unsigned* __restrict__ penc) {
    const int b  = blockIdx.y;
    const int tc = blockIdx.x;            // 16 chunks of 128 t's
    const int d  = threadIdx.x;           // 512
    const float* xb = x + ((size_t)b * T_ + (size_t)tc * 128) * D1_ + d;
    const int*   mb = mask + b * T_ + tc * 128;
    float m = -INFINITY;
    #pragma unroll 4
    for (int t = 0; t < 128; ++t) {
        float v = xb[(size_t)t * D1_];
        v += (mb[t] == 0) ? -1e6f : 0.0f;   // reference adds -1e6 to masked entries
        m = fmaxf(m, v);
    }
    atomicMax(&penc[b * D1_ + d], encf(m));
}

// ---- K2: Wt[d][k] = bf16(W_x[k][d])  (W_x = W rows 512..1023) ----
__global__ void k_transw(const float* __restrict__ W, bf16* __restrict__ Wt) {
    int i = blockIdx.x * blockDim.x + threadIdx.x;   // 512*512
    int d = i >> 9;
    int k = i & 511;
    Wt[(size_t)d * D1_ + k] = (bf16)W[(size_t)(512 + k) * D1_ + d];
}

// ---- K3: c[b][d] = p[b]@W_p + h[b]@W_h + bias ; also decode p into output ----
__global__ void k_combine(const unsigned* __restrict__ penc, const float* __restrict__ h,
                          const float* __restrict__ W, const float* __restrict__ bvec,
                          float* __restrict__ p_out, float* __restrict__ c_ws) {
    const int b = blockIdx.x;     // 32
    const int d = threadIdx.x;    // 512
    const unsigned* pe = penc + b * D1_;
    const float*    hb = h + b * D1_;
    float acc = bvec[d];
    for (int k = 0; k < D1_; ++k)
        acc += decf(pe[k]) * W[(size_t)k * D1_ + d];               // W_p rows 0..511
    for (int k = 0; k < D1_; ++k)
        acc += hb[k] * W[(size_t)(1024 + k) * D1_ + d];            // W_h rows 1024..1535
    c_ws[b * D1_ + d] = acc;
    p_out[b * D1_ + d] = decf(pe[d]);
}

// ---- K4: fused bf16-WMMA GEMM  score[b,t] = sum_d tanh(x@W_x + c[b])[d] * u[d] ----
#define LDA 520   // 512 + 8 bf16 pad -> row stride 1040B, conflict-free 16B ds reads

__global__ __launch_bounds__(512)
void k_gemm(const float* __restrict__ x, const bf16* __restrict__ Wt,
            const float* __restrict__ c_ws, const float* __restrict__ u,
            float* __restrict__ scores) {
    extern __shared__ char smem_raw[];
    bf16*  sx    = (bf16*)smem_raw;                       // [64][LDA]
    float* spart = (float*)(smem_raw + 64 * LDA * 2);     // [64 rows][16 waves]

    const int tid = threadIdx.x;
    const int b   = blockIdx.y;
    const int t0  = blockIdx.x * 64;

    // ---- stage 64x512 fp32 x-tile into LDS as bf16 ----
    const float* xb = x + ((size_t)b * T_ + t0) * D1_;
    #pragma unroll
    for (int i = 0; i < 16; ++i) {
        int e   = tid * 4 + i * 2048;        // element index in 64*512 tile
        int row = e >> 9;
        int col = e & 511;
        v4f f = *(const v4f*)(xb + (size_t)row * D1_ + col);
        v4bf o; o.x = (bf16)f.x; o.y = (bf16)f.y; o.z = (bf16)f.z; o.w = (bf16)f.w;
        *(v4bf*)(sx + row * LDA + col) = o;
    }
    __syncthreads();

    const int lane = tid & 31;
    const int wv   = tid >> 5;        // 0..15
    const int r    = lane & 15;       // row/col within 16
    const int hi   = lane >> 4;       // half-wave
    const int kb   = hi ? 8 : 0;      // K sub-offset per ISA A/B 16-bit layout

    // wave owns N-tiles {2*wv, 2*wv+1} for all 4 M-tiles
    const int d0 = 32 * wv + r;
    const int d1 = d0 + 16;
    const float c0 = c_ws[b * D1_ + d0], c1 = c_ws[b * D1_ + d1];
    const float u0 = u[d0],              u1 = u[d1];

    const bf16* w0 = Wt + (size_t)d0 * D1_ + kb;   // B fragment source rows (Wt = W_x^T)
    const bf16* w1 = Wt + (size_t)d1 * D1_ + kb;

    v8f acc[4][2] = {};

    for (int ks = 0; ks < 16; ++ks) {
        const int kk = ks * 32;
        union { v16bf v; v8bf h[2]; } tb0, tb1;
        tb0.h[0] = *(const v8bf*)(w0 + kk);
        tb0.h[1] = *(const v8bf*)(w0 + kk + 16);
        tb1.h[0] = *(const v8bf*)(w1 + kk);
        tb1.h[1] = *(const v8bf*)(w1 + kk + 16);

        v16bf afrag[4];
        #pragma unroll
        for (int mi = 0; mi < 4; ++mi) {
            union { v16bf v; v8bf h[2]; } ta;
            const bf16* ap = sx + (mi * 16 + r) * LDA + kk + kb;
            ta.h[0] = *(const v8bf*)(ap);
            ta.h[1] = *(const v8bf*)(ap + 16);
            afrag[mi] = ta.v;
        }
        #pragma unroll
        for (int mi = 0; mi < 4; ++mi) {
            acc[mi][0] = __builtin_amdgcn_wmma_f32_16x16x32_bf16(
                false, afrag[mi], false, tb0.v, (short)0, acc[mi][0], false, false);
            acc[mi][1] = __builtin_amdgcn_wmma_f32_16x16x32_bf16(
                false, afrag[mi], false, tb1.v, (short)0, acc[mi][1], false, false);
        }
    }

    // ---- epilogue: tanh(pre)*u, then reduce-scatter butterfly over 16 lanes ----
    // val[v], v = mi*8 + i : 32 partials per lane
    float val[32];
    #pragma unroll
    for (int mi = 0; mi < 4; ++mi)
        #pragma unroll
        for (int i = 0; i < 8; ++i)
            val[(mi << 3) | i] = fast_tanhf(acc[mi][0][i] + c0) * u0
                               + fast_tanhf(acc[mi][1][i] + c1) * u1;

    // step s: keep slots whose bit s matches lane bit s, add partner's copy.
    // after step s, slot j holds partial for v = (j << (s+1)) | lane[0:s].
    #pragma unroll
    for (int s = 0; s < 4; ++s) {
        const int bsel = (lane >> s) & 1;
        #pragma unroll
        for (int j = 0; j < (16 >> s); ++j) {
            float sel = bsel ? val[2 * j + 1] : val[2 * j];
            val[j] = sel + swz_xor(sel, s);
        }
    }
    // lane ends with sums for v = (lane&15) and (lane&15)+16
    {
        const int ll   = lane & 15;
        const int row0 = ((ll >> 3) & 1) * 16 + 8 * hi + (ll & 7);   // mi in {0,1}
        spart[row0 * 16 + wv]        = val[0];                        // v = ll
        spart[(row0 + 32) * 16 + wv] = val[1];                        // v = ll+16 -> mi+2
    }
    __syncthreads();
    if (tid < 64) {
        float s = 0.0f;
        #pragma unroll
        for (int j = 0; j < 16; ++j) s += spart[tid * 16 + j];   // fixed order: deterministic
        scores[(size_t)b * T_ + t0 + tid] = s;
    }
}

// ---- K5: masked softmax over T per batch ----
__global__ void k_softmax(const float* __restrict__ scores, const int* __restrict__ mask,
                          float* __restrict__ e_out) {
    const int b = blockIdx.x;
    const int tid = threadIdx.x;      // 256
    __shared__ float red[256];
    float ev[8];
    float s = 0.0f;
    #pragma unroll
    for (int i = 0; i < 8; ++i) {
        int t = tid + i * 256;
        float e = expf(scores[b * T_ + t]) * (float)mask[b * T_ + t];
        ev[i] = e; s += e;
    }
    red[tid] = s;
    __syncthreads();
    for (int off = 128; off > 0; off >>= 1) {
        if (tid < off) red[tid] += red[tid + off];
        __syncthreads();
    }
    const float denom = red[0] + 1e-7f;
    #pragma unroll
    for (int i = 0; i < 8; ++i)
        e_out[b * T_ + tid + i * 256] = ev[i] / denom;
}

extern "C" void kernel_launch(void* const* d_in, const int* in_sizes, int n_in,
                              void* d_out, int out_size, void* d_ws, size_t ws_size,
                              hipStream_t stream) {
    const float* x    = (const float*)d_in[0];
    const float* h    = (const float*)d_in[1];
    const int*   mask = (const int*)  d_in[2];
    const float* W    = (const float*)d_in[3];
    const float* u    = (const float*)d_in[4];
    const float* bvec = (const float*)d_in[5];

    float* p_out = (float*)d_out;                 // [32,512]
    float* e_out = (float*)d_out + B_ * D1_;      // [32,2048]

    char* ws = (char*)d_ws;
    bf16*     Wt     = (bf16*)ws;                                   // 512*512*2 = 524288 B
    unsigned* penc   = (unsigned*)(ws + 524288);                    // 65536 B
    float*    c_ws   = (float*)(ws + 524288 + 65536);               // 65536 B
    float*    scores = (float*)(ws + 524288 + 65536 + 65536);       // 262144 B

    k_init<<<dim3((B_ * D1_ + 255) / 256), dim3(256), 0, stream>>>(penc);
    k_transw<<<dim3(D1_ * D1_ / 256), dim3(256), 0, stream>>>(W, Wt);
    k_maxpool<<<dim3(16, B_), dim3(512), 0, stream>>>(x, mask, penc);
    k_combine<<<dim3(B_), dim3(512), 0, stream>>>(penc, h, W, bvec, p_out, c_ws);

    const size_t smem = (size_t)64 * LDA * 2 + (size_t)64 * 16 * 4;   // 70656 B
    k_gemm<<<dim3(T_ / 64, B_), dim3(512), smem, stream>>>(x, Wt, c_ws, u, scores);

    k_softmax<<<dim3(B_), dim3(256), 0, stream>>>(scores, mask, e_out);
}
